// LVCBlock_78245714199030
// MI455X (gfx1250) — compile-verified, matrix operand order checked by reading, not run on hardware
//
#include <hip/hip_runtime.h>
#include <math.h>

typedef _Float16 f16;
typedef __attribute__((ext_vector_type(16))) _Float16 v16h;
typedef __attribute__((ext_vector_type(8)))  float    v8f;

#define B_      8
#define T_IN    4096
#define C_IN_   32
#define T_UP    32768
#define CLEN    128
#define CCH     100
#define HID_    64
#define C_OUT_  64
#define KCOLS   24576   // L_W
#define KDIM    192     // 3*HID im2col depth
#define M_ROWS  1024    // B*COND_LEN
#define SLOPE_  0.2f

__device__ __forceinline__ float lrelu_f(float v) { return v >= 0.f ? v : SLOPE_ * v; }

// -------------------------------------------------------------------------
// Kernel 1: conditioning network.  h = lrelu(conv5(c)) then 3 residual blocks.
// one block per batch, h kept in LDS.
// -------------------------------------------------------------------------
__global__ void cond_net(const float* __restrict__ c,
                         const float* __restrict__ w_in, const float* __restrict__ b_in,
                         const float* __restrict__ rw1,  const float* __restrict__ rb1,
                         const float* __restrict__ rw2,  const float* __restrict__ rb2,
                         float* __restrict__ h_out) {
    __shared__ float sh[CLEN * HID_];   // 32 KB
    __shared__ f16   sr[CLEN * HID_];   // 16 KB
    const int b = blockIdx.x, tid = threadIdx.x;

    for (int idx = tid; idx < CLEN * HID_; idx += 256) {
        int t = idx >> 6, o = idx & 63;
        float acc = b_in[o];
        for (int j = 0; j < 5; ++j) {
            int tt = t + j - 2;
            if (tt < 0 || tt >= CLEN) continue;
            const float* crow = c + (b * CLEN + tt) * CCH;
            const float* w    = w_in + (j * CCH) * HID_ + o;
            for (int cc = 0; cc < CCH; ++cc) acc += crow[cc] * w[cc * HID_];
        }
        sh[idx] = lrelu_f(acc);
    }
    __syncthreads();

    for (int i = 0; i < 3; ++i) {
        for (int idx = tid; idx < CLEN * HID_; idx += 256) {
            int t = idx >> 6, o = idx & 63;
            float acc = rb1[i * HID_ + o];
            for (int j = 0; j < 3; ++j) {
                int tt = t + j - 1;
                if (tt < 0 || tt >= CLEN) continue;
                const float* hr = sh + tt * HID_;
                const float* w  = rw1 + ((i * 3 + j) * HID_) * HID_ + o;
                for (int cc = 0; cc < HID_; ++cc) acc += hr[cc] * w[cc * HID_];
            }
            sr[idx] = (f16)lrelu_f(acc);
        }
        __syncthreads();
        for (int idx = tid; idx < CLEN * HID_; idx += 256) {
            int t = idx >> 6, o = idx & 63;
            float acc = rb2[i * HID_ + o];
            for (int j = 0; j < 3; ++j) {
                int tt = t + j - 1;
                if (tt < 0 || tt >= CLEN) continue;
                const f16*   rr = sr + tt * HID_;
                const float* w  = rw2 + ((i * 3 + j) * HID_) * HID_ + o;
                for (int cc = 0; cc < HID_; ++cc) acc += (float)rr[cc] * w[cc * HID_];
            }
            sh[idx] += lrelu_f(acc);
        }
        __syncthreads();
    }
    for (int idx = tid; idx < CLEN * HID_; idx += 256)
        h_out[b * (CLEN * HID_) + idx] = sh[idx];
}

// -------------------------------------------------------------------------
// Kernel 2: pack f16 operands.  A = im2col(h) [1024][192];
// Wt_k = w_k^T [24576][192];  Wt_b = w_b^T [256][192].
// -------------------------------------------------------------------------
__global__ void pack(const float* __restrict__ h, const float* __restrict__ w_k,
                     const float* __restrict__ w_b,
                     f16* __restrict__ A, f16* __restrict__ Wtk, f16* __restrict__ Wtb) {
    size_t idx = (size_t)blockIdx.x * 256 + threadIdx.x;
    if (idx < (size_t)M_ROWS * KDIM) {
        int m = idx / KDIM, k = idx % KDIM;
        int j = k / HID_, cc = k % HID_;
        int b = m / CLEN, l = m % CLEN;
        int t = l + j - 1;
        float v = (t >= 0 && t < CLEN) ? h[(b * CLEN + t) * HID_ + cc] : 0.f;
        A[idx] = (f16)v;
    }
    if (idx < (size_t)KCOLS * KDIM) {
        int n = idx / KDIM, k = idx % KDIM;
        Wtk[idx] = (f16)w_k[(size_t)k * KCOLS + n];
    }
    if (idx < (size_t)256 * KDIM) {
        int n = idx / KDIM, k = idx % KDIM;
        Wtb[idx] = (f16)w_b[k * 256 + n];
    }
}

// -------------------------------------------------------------------------
// Kernel 3: kernel-predictor GEMM with WMMA f32<-f16 16x16x32.
// D[1024][24576] (f16) = A[1024][192] * Wt_k^T + b_k.
// block = 8 waves; each wave owns one 16x16 tile; grid = (24576/128, 1024/16).
// -------------------------------------------------------------------------
__global__ void gemm_kernels(const f16* __restrict__ A, const f16* __restrict__ Wt,
                             const float* __restrict__ bk, f16* __restrict__ D) {
    const int wave = threadIdx.x >> 5, lane = threadIdx.x & 31;
    const int hi = lane >> 4, ln = lane & 15;
    const int m0 = blockIdx.y * 16;
    const int n0 = (blockIdx.x * 8 + wave) * 16;
    const int mr = m0 + ln;    // A row for this lane
    const int nc = n0 + ln;    // output / B column for this lane

    v8f acc = {};
    const f16* Arow = A  + mr * KDIM;
    const f16* Brow = Wt + nc * KDIM;
    for (int ks = 0; ks < 6; ++ks) {
        const int k0 = ks * 32;
        __builtin_prefetch(Brow + k0 + 32, 0, 1);
        v16h av, bv;
#pragma unroll
        for (int i = 0; i < 8; ++i) {
            // ISA 16-bit A layout: VGPR i holds K pair; lanes 16-31 offset K by +8
            int kb = ((i < 4) ? 2 * i : 16 + 2 * (i - 4)) + (hi ? 8 : 0) + k0;
            av[2 * i]     = Arow[kb];
            av[2 * i + 1] = Arow[kb + 1];
            bv[2 * i]     = Brow[kb];
            bv[2 * i + 1] = Brow[kb + 1];
        }
        acc = __builtin_amdgcn_wmma_f32_16x16x32_f16(false, av, false, bv,
                                                     (short)0, acc, false, false);
    }
    const float bias = bk[nc];
#pragma unroll
    for (int v = 0; v < 8; ++v) {
        int mrow = m0 + v + 8 * hi;             // ISA C/D layout: VGPR v -> M = v + 8*hi
        D[(size_t)mrow * KCOLS + nc] = (f16)(acc[v] + bias);
    }
}

// -------------------------------------------------------------------------
// Kernel 4: bias-predictor GEMM (tiny) -> biases f32 [1024][256]
// -------------------------------------------------------------------------
__global__ void bias_gemm(const f16* __restrict__ A, const float* __restrict__ w_b,
                          const float* __restrict__ b_b, float* __restrict__ out) {
    int idx = blockIdx.x * 256 + threadIdx.x;   // 1024*256 exact
    int m = idx >> 8, o = idx & 255;
    float acc = b_b[o];
    const f16* a = A + m * KDIM;
    for (int k = 0; k < KDIM; ++k) acc += (float)a[k] * w_b[k * 256 + o];
    out[idx] = acc;
}

// -------------------------------------------------------------------------
// Kernel 5: conv_transpose(lrelu(x), w_ct, stride 8, SAME) + b_ct -> xu
// SAME transpose padding: pad_a = pad_b = 11; each output sums taps k0, k0+8.
// -------------------------------------------------------------------------
__global__ void conv_transpose_k(const float* __restrict__ x, const float* __restrict__ w_ct,
                                 const float* __restrict__ b_ct, float* __restrict__ xu) {
    int idx = blockIdx.x * 256 + threadIdx.x;   // B*T_UP*C_IN exact
    int co = idx & 31, t = (idx >> 5) & (T_UP - 1), b = idx >> 20;
    int k0 = ((11 - t) % 8 + 8) % 8;
    float acc = b_ct[co];
#pragma unroll
    for (int p = 0; p < 2; ++p) {
        int k = k0 + p * 8;
        int mnum = t + k - 11;
        if (mnum < 0) continue;
        int mi = mnum >> 3;
        if (mi >= T_IN) continue;
        const float* xr = x + ((size_t)b * T_IN + mi) * C_IN_;
        const float* w  = w_ct + (k * C_IN_) * C_IN_ + co;
        for (int cc = 0; cc < C_IN_; ++cc) acc += lrelu_f(xr[cc]) * w[cc * C_IN_];
    }
    xu[idx] = acc;
}

// -------------------------------------------------------------------------
// Kernel 6: one LVC layer.  Block = (b, l) chunk of 256 samples, 256 threads.
//  phase 0: stage lrelu(xu) halo (f16), transposed LVC kernel, transposed cw
//  phase 1: dilated 3-tap conv as WMMA (M=272 pad, N=32, K=96) -> out1 f16
//  phase 2: LVC einsum as WMMA: 16 M-tiles x 4 N-tiles, K=96 (3 WMMA steps)
//  phase 3: xu_dst = xu_src + sigmoid(out[:32]) * tanh(out[32:])
// LDS region A (32KB): xu stage [<=312][32] f16 (19.9KB) + cwT at +22528
//                      (6KB); later reused as gate [256][64] f16 (32KB).
// out1 [258][32] f16 = 16.5KB; bT [64][96] f16 = 12KB.  Total 61.5KB.
// -------------------------------------------------------------------------
#define SM_GATE_OFF 0
#define SM_CWT_OFF  22528
#define SM_OUT1_OFF 32768
#define SM_BT_OFF   (32768 + 16512)
#define SM_BYTES    (32768 + 16512 + 12288)

__global__ void lvc_layer(const float* __restrict__ src, float* __restrict__ dst,
                          const f16* __restrict__ kernels, const float* __restrict__ biases,
                          const float* __restrict__ cw, const float* __restrict__ cb,
                          int layer, int d) {
    __shared__ __align__(16) char smem[SM_BYTES];
    f16* xu_s = (f16*)(smem + SM_GATE_OFF);   // [R][32], R = 258+2d <= 312
    f16* gate = (f16*)(smem + SM_GATE_OFF);   // [256][64], reuses region A
    f16* cwT  = (f16*)(smem + SM_CWT_OFF);    // [32][96]  dilated-conv weights^T
    f16* out1 = (f16*)(smem + SM_OUT1_OFF);   // [258][32]
    f16* bT   = (f16*)(smem + SM_BT_OFF);     // [64][96]  LVC kernel^T (kappa pairs contiguous)

    const int tid = threadIdx.x;
    const int lane = tid & 31, wave = tid >> 5, hi = lane >> 4, ln = lane & 15;
    const int m = blockIdx.x;                 // b*128 + l
    const int b = m >> 7, l = m & 127;
    const int t0 = l * 256;
    const int R = 258 + 2 * d;
    const int tbase = t0 - 1 - d;

    // phase 0: stage inputs
    for (int idx = tid; idx < R * 32; idx += 256) {
        int r = idx >> 5, c = idx & 31;
        int t = tbase + r;
        float v = (t >= 0 && t < T_UP) ? src[((size_t)b * T_UP + t) * C_IN_ + c] : 0.f;
        xu_s[idx] = (f16)lrelu_f(v);
    }
    for (int idx = tid; idx < 32 * 96; idx += 256) {
        int o = idx / 96, kap = idx % 96;
        int j = kap >> 5, c = kap & 31;
        cwT[idx] = (f16)cw[((layer * 3 + j) * 32 + c) * 32 + o];   // cw[i][j][c][o]
    }
    const f16* kerrow = kernels + (size_t)m * KCOLS + layer * 6144;
    for (int idx = tid; idx < 64 * 96; idx += 256) {
        int o = idx / 96, kap = idx % 96;
        int k = kap >> 5, c = kap & 31;
        bT[o * 96 + kap] = kerrow[c * 192 + o * 3 + k];            // ker[c][o][k]
    }
    __syncthreads();

    // phase 1: dilated conv as WMMA.  A[s][kappa] = xu_s[s + (kappa/32)*d][kappa%32]
    // 17 M-tiles (272 rows, 258 valid) x 2 N-tiles; wave-uniform tile loop.
    for (int tile = wave; tile < 34; tile += 8) {
        int mtile = tile >> 1, ntile = tile & 1;
        int srow = mtile * 16 + ln;
        v8f acc = {};
        for (int ks = 0; ks < 3; ++ks) {
            v16h av, bv;
#pragma unroll
            for (int i = 0; i < 8; ++i) {
                int kb = ((i < 4) ? 2 * i : 16 + 2 * (i - 4)) + (hi ? 8 : 0) + ks * 32;
                int j = kb >> 5, cc = kb & 31;
                const f16* ap = xu_s + (srow + j * d) * 32 + cc;
                av[2 * i]     = ap[0];
                av[2 * i + 1] = ap[1];
                const f16* bp = cwT + (ntile * 16 + ln) * 96 + kb;
                bv[2 * i]     = bp[0];
                bv[2 * i + 1] = bp[1];
            }
            acc = __builtin_amdgcn_wmma_f32_16x16x32_f16(false, av, false, bv,
                                                         (short)0, acc, false, false);
        }
        int o = ntile * 16 + ln;
        float cbv = cb[layer * 32 + o];
#pragma unroll
        for (int v = 0; v < 8; ++v) {
            int s = mtile * 16 + v + 8 * hi;
            if (s < 258) {
                int t = t0 - 1 + s;
                float val = (t >= 0 && t < T_UP) ? lrelu_f(acc[v] + cbv) : 0.f;
                out1[s * 32 + o] = (f16)val;   // zero rows = frame padding
            }
        }
    }
    __syncthreads();   // out1 ready; region A (xu_s/cwT) now dead -> gate

    // phase 2: LVC einsum as WMMA. A[s][kappa]=out1[s + kappa/32][kappa%32], B=bT.
    for (int q = 0; q < 8; ++q) {
        int tileid = wave + q * 8;
        int mtile = tileid >> 2, ntile = tileid & 3;
        int srow = mtile * 16 + ln;
        v8f acc = {};
        for (int ks = 0; ks < 3; ++ks) {
            v16h av, bv;
#pragma unroll
            for (int i = 0; i < 8; ++i) {
                int kb = ((i < 4) ? 2 * i : 16 + 2 * (i - 4)) + (hi ? 8 : 0) + ks * 32;
                int kk = kb >> 5, cc = kb & 31;           // kappa -> (tap k, channel c)
                const f16* ap = out1 + (srow + kk) * 32 + cc;
                av[2 * i]     = ap[0];
                av[2 * i + 1] = ap[1];
                const f16* bp = bT + (ntile * 16 + ln) * 96 + kb;
                bv[2 * i]     = bp[0];
                bv[2 * i + 1] = bp[1];
            }
            acc = __builtin_amdgcn_wmma_f32_16x16x32_f16(false, av, false, bv,
                                                         (short)0, acc, false, false);
        }
        int o = ntile * 16 + ln;
        float bias = biases[(size_t)m * 256 + layer * 64 + o];
#pragma unroll
        for (int v = 0; v < 8; ++v) {
            int s = mtile * 16 + v + 8 * hi;
            gate[s * 64 + o] = (f16)(acc[v] + bias);
        }
    }
    __syncthreads();

    // phase 3: gated residual update
    for (int idx = tid; idx < 256 * 32; idx += 256) {
        int s = idx >> 5, c = idx & 31;
        float g1 = (float)gate[s * 64 + c];
        float g2 = (float)gate[s * 64 + 32 + c];
        float sig = 1.f / (1.f + __expf(-g1));
        float th  = tanhf(g2);
        size_t off = ((size_t)b * T_UP + (t0 + s)) * C_IN_ + c;
        dst[off] = src[off] + sig * th;
    }
}

// -------------------------------------------------------------------------
extern "C" void kernel_launch(void* const* d_in, const int* in_sizes, int n_in,
                              void* d_out, int out_size, void* d_ws, size_t ws_size,
                              hipStream_t stream) {
    const float* x    = (const float*)d_in[0];
    const float* c    = (const float*)d_in[1];
    const float* w_ct = (const float*)d_in[2];
    const float* b_ct = (const float*)d_in[3];
    const float* w_in = (const float*)d_in[4];
    const float* b_in = (const float*)d_in[5];
    const float* rw1  = (const float*)d_in[6];
    const float* rb1  = (const float*)d_in[7];
    const float* rw2  = (const float*)d_in[8];
    const float* rb2  = (const float*)d_in[9];
    const float* w_k  = (const float*)d_in[10];
    const float* b_k  = (const float*)d_in[11];
    const float* w_b  = (const float*)d_in[12];
    const float* b_b  = (const float*)d_in[13];
    const float* cw   = (const float*)d_in[14];
    const float* cb   = (const float*)d_in[15];
    float* out = (float*)d_out;

    char* ws = (char*)d_ws;
    size_t off = 0;
    auto alloc = [&](size_t bytes) { size_t o = off; off = (off + bytes + 255) & ~(size_t)255; return o; };
    float* h_buf   = (float*)(ws + alloc((size_t)B_ * CLEN * HID_ * 4));       // 256 KB
    f16*   A_buf   = (f16*)  (ws + alloc((size_t)M_ROWS * KDIM * 2));          // 384 KB
    f16*   Wtk     = (f16*)  (ws + alloc((size_t)KCOLS * KDIM * 2));           // 9 MB
    f16*   Wtb     = (f16*)  (ws + alloc((size_t)256 * KDIM * 2));             // 96 KB
    f16*   kerbuf  = (f16*)  (ws + alloc((size_t)M_ROWS * KCOLS * 2));         // 48 MB
    float* biasbuf = (float*)(ws + alloc((size_t)M_ROWS * 256 * 4));           // 1 MB
    float* xu_tmp  = (float*)(ws + alloc((size_t)B_ * T_UP * C_IN_ * 4));      // 32 MB

    cond_net<<<B_, 256, 0, stream>>>(c, w_in, b_in, rw1, rb1, rw2, rb2, h_buf);

    {
        size_t tot = (size_t)KCOLS * KDIM;   // largest of the three pack jobs
        int grid = (int)((tot + 255) / 256);
        pack<<<grid, 256, 0, stream>>>(h_buf, w_k, w_b, A_buf, Wtk, Wtb);
    }

    gemm_kernels<<<dim3(KCOLS / 128, M_ROWS / 16), 256, 0, stream>>>(A_buf, Wtk, b_k, kerbuf);

    bias_gemm<<<(M_ROWS * 256) / 256, 256, 0, stream>>>(A_buf, w_b, b_b, biasbuf);

    conv_transpose_k<<<(B_ * T_UP * C_IN_) / 256, 256, 0, stream>>>(x, w_ct, b_ct, out);

    // ping-pong: out -> tmp -> out -> tmp -> out
    lvc_layer<<<M_ROWS, 256, 0, stream>>>(out,    xu_tmp, kerbuf, biasbuf, cw, cb, 0, 1);
    lvc_layer<<<M_ROWS, 256, 0, stream>>>(xu_tmp, out,    kerbuf, biasbuf, cw, cb, 1, 3);
    lvc_layer<<<M_ROWS, 256, 0, stream>>>(out,    xu_tmp, kerbuf, biasbuf, cw, cb, 2, 9);
    lvc_layer<<<M_ROWS, 256, 0, stream>>>(xu_tmp, out,    kerbuf, biasbuf, cw, cb, 3, 27);
}
